// CausalSelfAttention_17867063951413
// MI455X (gfx1250) — matrix operand order repaired and checked by reading, not compiled, not run on hardware
//
#include <hip/hip_runtime.h>

typedef __attribute__((ext_vector_type(16))) __bf16 v16bf;
typedef __attribute__((ext_vector_type(8)))  __bf16 v8bf;
typedef __attribute__((ext_vector_type(8)))  float  v8f;
typedef int v4i_vec __attribute__((vector_size(16)));  // matches async-LDS builtin param

union FragA { v16bf f; v8bf h[2]; };

#define WMMA_BF16(a, b, c) \
  __builtin_amdgcn_wmma_f32_16x16x32_bf16(false, (a), false, (b), (short)0, (c), false, false)

#if defined(__HIP_DEVICE_COMPILE__) && __has_builtin(__builtin_amdgcn_global_load_async_to_lds_b128)
#define HAS_ASYNC_LDS 1
#else
#define HAS_ASYNC_LDS 0
#endif

// 16-byte global -> LDS copy: async (ASYNCcnt-tracked) when available.
__device__ __forceinline__ void cp16_g2l(__bf16* ldst, const __bf16* gsrc) {
#if HAS_ASYNC_LDS
  __builtin_amdgcn_global_load_async_to_lds_b128(
      (__attribute__((address_space(1))) v4i_vec*)gsrc,
      (__attribute__((address_space(3))) v4i_vec*)ldst, /*offset=*/0, /*cpol=*/0);
#else
  *(v8bf*)ldst = *(const v8bf*)gsrc;
#endif
}

__device__ __forceinline__ void cp_async_wait() {
#if HAS_ASYNC_LDS
#if __has_builtin(__builtin_amdgcn_s_wait_asynccnt)
  __builtin_amdgcn_s_wait_asynccnt(0);
#else
  asm volatile("s_wait_asynccnt 0" ::: "memory");
#endif
#endif
}

// Problem constants: B=2, T=3072, C=1024, H=16, d=64, t=1024
// M = B*T = 6144 rows for all projection GEMMs, N = K = 1024.

__global__ __launch_bounds__(256) void cvt_f32_bf16(const float* __restrict__ in,
                                                    __bf16* __restrict__ out, int n) {
  int i = blockIdx.x * 256 + threadIdx.x;
  if (i < n) out[i] = (__bf16)in[i];
}

// out = A[M,K] @ W[N,K]^T + bias, bf16 inputs, f32 accumulate via WMMA.
// Block: 256 threads = 8 waves; per-wave tile 32x64; block tile 256x64.
// W tile (64 cols x 32 K) staged in double-buffered LDS via async loads.
// mode 0: obf in [B,H,T,d]   (q, k)
// mode 1: obf in [B,H,d,T]   (v transposed for the P@V B-fragments)
// mode 2: of32 in [B,T,C]    (final projection)
__global__ __launch_bounds__(256) void gemm_bf16_wmma(
    const __bf16* __restrict__ A,    // [6144, 1024] row-major
    const __bf16* __restrict__ W,    // [1024, 1024] row-major (Linear weight)
    const float*  __restrict__ bias, // [1024]
    __bf16* __restrict__ obf,
    float*  __restrict__ of32,
    int mode, float scale) {
  __shared__ __align__(32) __bf16 wtile[2][64][32];  // 2 x 4KB double buffer

  const int tid  = threadIdx.x;
  const int lane = tid & 31;
  const int wv   = tid >> 5;
  const int half = lane >> 4;   // 0: lanes 0-15, 1: lanes 16-31
  const int l16  = lane & 15;
  const int m0 = blockIdx.y * 256 + wv * 32;  // 8 waves * 32 rows
  const int n0 = blockIdx.x * 64;

  // Async staging assignment: thread -> (row, 8-element K chunk) of the W tile
  const int srow = tid >> 2;          // 0..63
  const int skc  = (tid & 3) * 8;     // 0,8,16,24
  const __bf16* wsrc = W + (size_t)(n0 + srow) * 1024 + skc;

  v8f acc[2][4] = {};

  // A fragment: lane<16 holds K {0..15}; lane>=16 holds {16..31}
  const __bf16* arow0 = A + (size_t)(m0 + l16) * 1024 + half * 16;
  const __bf16* arow1 = A + (size_t)(m0 + 16 + l16) * 1024 + half * 16;

  // Prologue: stage first K tile
  cp16_g2l(&wtile[0][srow][skc], wsrc);

  int buf = 0;
  for (int kb = 0; kb < 1024; kb += 32) {
    cp_async_wait();
    __syncthreads();
    if (kb + 32 < 1024) cp16_g2l(&wtile[buf ^ 1][srow][skc], wsrc + kb + 32);

    FragA a0, a1;
    a0.h[0] = *(const v8bf*)(arow0 + kb);
    a0.h[1] = *(const v8bf*)(arow0 + kb + 8);
    a1.h[0] = *(const v8bf*)(arow1 + kb);
    a1.h[1] = *(const v8bf*)(arow1 + kb + 8);

#pragma unroll
    for (int j = 0; j < 4; ++j) {
      // B fragment: lane holds column n (= j*16+l16), 16 consecutive K
      v16bf b = *(const v16bf*)&wtile[buf][j * 16 + l16][half * 16];
      acc[0][j] = WMMA_BF16(a0.f, b, acc[0][j]);
      acc[1][j] = WMMA_BF16(a1.f, b, acc[1][j]);
    }
    buf ^= 1;
  }

  // C/D layout: VGPR i, lane L -> row +i+(L>=16)*8, col tile*16 + L%16
#pragma unroll
  for (int mi = 0; mi < 2; ++mi) {
#pragma unroll
    for (int j = 0; j < 4; ++j) {
      const int n = n0 + j * 16 + l16;
      const float bv = bias[n];
#pragma unroll
      for (int i = 0; i < 8; ++i) {
        const int r = m0 + mi * 16 + i + half * 8;   // global row in [0, 6144)
        const float val = (acc[mi][j][i] + bv) * scale;
        if (mode == 2) {
          of32[(size_t)r * 1024 + n] = val;
        } else {
          const int bidx = r / 3072;
          const int tq   = r % 3072;
          const int hh   = n >> 6;
          const int dc   = n & 63;
          if (mode == 0) {
            obf[(((size_t)bidx * 16 + hh) * 3072 + tq) * 64 + dc] = (__bf16)val;
          } else {
            obf[(((size_t)bidx * 16 + hh) * 64 + dc) * 3072 + tq] = (__bf16)val;
          }
        }
      }
    }
  }
}

// Flash attention with block-repeated causal mask: allowed iff (tq%1024) >= (tk%1024).
// q pre-scaled by 1/sqrt(64). One wave per 16 query rows, kv blocks of 32.
__global__ __launch_bounds__(128) void attn_wmma(
    const __bf16* __restrict__ q,   // [B*H, T, 64]
    const __bf16* __restrict__ k,   // [B*H, T, 64]
    const __bf16* __restrict__ vT,  // [B*H, 64, T]
    __bf16* __restrict__ y) {       // [B, T, C]
  __shared__ __align__(32) __bf16 lds[4][16][32];  // per-wave P staging
  const int lane = threadIdx.x & 31;
  const int wv   = threadIdx.x >> 5;
  const int half = lane >> 4;
  const int l16  = lane & 15;
  const int bh = blockIdx.y;
  const int b  = bh >> 4;
  const int h  = bh & 15;
  const int q0 = blockIdx.x * 64 + wv * 16;
  const int qm0 = q0 & 1023;

  const __bf16* qb  = q  + (size_t)bh * 3072 * 64;
  const __bf16* kbp = k  + (size_t)bh * 3072 * 64;
  const __bf16* vb  = vT + (size_t)bh * 64 * 3072;

  // Preload q A-fragments for K=0..63 (two 16x32 fragments)
  FragA aq0, aq1;
  {
    const __bf16* qr = qb + (size_t)(q0 + l16) * 64 + half * 16;
    aq0.h[0] = *(const v8bf*)(qr + 0);
    aq0.h[1] = *(const v8bf*)(qr + 8);
    aq1.h[0] = *(const v8bf*)(qr + 32);
    aq1.h[1] = *(const v8bf*)(qr + 40);
  }

  v8f accy[4] = {};
  float mrow[8], lrow[8];
#pragma unroll
  for (int i = 0; i < 8; ++i) { mrow[i] = -1e30f; lrow[i] = 0.f; }

  for (int kc = 0; kc < 3072; kc += 32) {
    const int km0 = kc & 1023;
    if (km0 > qm0 + 15) continue;              // fully masked kv block
    const bool full = (km0 + 31) <= qm0;       // no per-element mask needed

    // S tile (16 x 32): two 16-key column tiles, K=64 in two wmma steps
    v8f s0 = {}, s1 = {};
    {
      const __bf16* kr0 = kbp + (size_t)(kc + l16) * 64 + half * 16;
      const __bf16* kr1 = kbp + (size_t)(kc + 16 + l16) * 64 + half * 16;
      v16bf bk00 = *(const v16bf*)(kr0);
      v16bf bk01 = *(const v16bf*)(kr0 + 32);
      v16bf bk10 = *(const v16bf*)(kr1);
      v16bf bk11 = *(const v16bf*)(kr1 + 32);
      s0 = WMMA_BF16(aq0.f, bk00, s0);
      s0 = WMMA_BF16(aq1.f, bk01, s0);
      s1 = WMMA_BF16(aq0.f, bk10, s1);
      s1 = WMMA_BF16(aq1.f, bk11, s1);
    }

    if (!full) {
#pragma unroll
      for (int i = 0; i < 8; ++i) {
        const int qm = qm0 + i + half * 8;
        if (qm < km0 + l16)      s0[i] = -1e30f;
        if (qm < km0 + 16 + l16) s1[i] = -1e30f;
      }
    }

    // Online softmax: row stats live per-half-wave (row = i + half*8)
#pragma unroll
    for (int i = 0; i < 8; ++i) {
      float bm = fmaxf(s0[i], s1[i]);
#pragma unroll
      for (int off = 8; off >= 1; off >>= 1)
        bm = fmaxf(bm, __shfl_xor(bm, off, 32));
      const float nm = fmaxf(mrow[i], bm);
      const float alpha = __expf(mrow[i] - nm);
      mrow[i] = nm;
      const float p0 = __expf(s0[i] - nm);
      const float p1 = __expf(s1[i] - nm);
      float rs = p0 + p1;
#pragma unroll
      for (int off = 8; off >= 1; off >>= 1)
        rs += __shfl_xor(rs, off, 32);
      lrow[i] = lrow[i] * alpha + rs;
      accy[0][i] *= alpha; accy[1][i] *= alpha;
      accy[2][i] *= alpha; accy[3][i] *= alpha;
      lds[wv][i + half * 8][l16]      = (__bf16)p0;
      lds[wv][i + half * 8][16 + l16] = (__bf16)p1;
    }

    // Re-materialize P as an A-fragment from LDS (same-wave DS ops are in-order)
    FragA ap;
    ap.h[0] = *(const v8bf*)&lds[wv][l16][half * 16];
    ap.h[1] = *(const v8bf*)&lds[wv][l16][half * 16 + 8];

    // y += P @ V : vT rows give contiguous K (=tk) per output column d
#pragma unroll
    for (int c = 0; c < 4; ++c) {
      const int dd = c * 16 + l16;
      v16bf bvf = *(const v16bf*)(vb + (size_t)dd * 3072 + kc + half * 16);
      accy[c] = WMMA_BF16(ap.f, bvf, accy[c]);
    }
  }

#pragma unroll
  for (int c = 0; c < 4; ++c) {
    const int dd = c * 16 + l16;
#pragma unroll
    for (int i = 0; i < 8; ++i) {
      const int tq = q0 + i + half * 8;
      y[((size_t)b * 3072 + tq) * 1024 + h * 64 + dd] = (__bf16)(accy[c][i] / lrow[i]);
    }
  }
}

extern "C" void kernel_launch(void* const* d_in, const int* in_sizes, int n_in,
                              void* d_out, int out_size, void* d_ws, size_t ws_size,
                              hipStream_t stream) {
  const float* x  = (const float*)d_in[0];
  const float* Wq = (const float*)d_in[1];
  const float* bq = (const float*)d_in[2];
  const float* Wk = (const float*)d_in[3];
  const float* bk = (const float*)d_in[4];
  const float* Wv = (const float*)d_in[5];
  const float* bv = (const float*)d_in[6];
  const float* Wp = (const float*)d_in[7];
  const float* bp = (const float*)d_in[8];
  float* out = (float*)d_out;

  const size_t nX = (size_t)6144 * 1024;   // B*T*C
  const size_t nW = (size_t)1024 * 1024;   // C*C
  __bf16* ws  = (__bf16*)d_ws;
  __bf16* xb  = ws;  ws += nX;
  __bf16* wqb = ws;  ws += nW;
  __bf16* wkb = ws;  ws += nW;
  __bf16* wvb = ws;  ws += nW;
  __bf16* wpb = ws;  ws += nW;
  __bf16* qb  = ws;  ws += nX;
  __bf16* kb  = ws;  ws += nX;
  __bf16* vtb = ws;  ws += nX;
  __bf16* yb  = ws;  ws += nX;

  cvt_f32_bf16<<<(int)((nX + 255) / 256), 256, 0, stream>>>(x,  xb,  (int)nX);
  cvt_f32_bf16<<<(int)((nW + 255) / 256), 256, 0, stream>>>(Wq, wqb, (int)nW);
  cvt_f32_bf16<<<(int)((nW + 255) / 256), 256, 0, stream>>>(Wk, wkb, (int)nW);
  cvt_f32_bf16<<<(int)((nW + 255) / 256), 256, 0, stream>>>(Wv, wvb, (int)nW);
  cvt_f32_bf16<<<(int)((nW + 255) / 256), 256, 0, stream>>>(Wp, wpb, (int)nW);

  dim3 gg(16, 24);  // N/64, M/256
  gemm_bf16_wmma<<<gg, 256, 0, stream>>>(xb, wqb, bq, qb,  nullptr, 0, 0.125f);
  gemm_bf16_wmma<<<gg, 256, 0, stream>>>(xb, wkb, bk, kb,  nullptr, 0, 1.0f);
  gemm_bf16_wmma<<<gg, 256, 0, stream>>>(xb, wvb, bv, vtb, nullptr, 1, 1.0f);

  attn_wmma<<<dim3(48, 32), 128, 0, stream>>>(qb, kb, vtb, yb);

  gemm_bf16_wmma<<<gg, 256, 0, stream>>>(yb, wpb, bp, nullptr, out, 2, 1.0f);
}
